// CRF_25082609009363
// MI455X (gfx1250) — compile-verified, matrix operand order checked by reading, not run on hardware
//
#include <hip/hip_runtime.h>

// CRF forward scan for MI455X (gfx1250, wave32).
// Per-step logsumexp recurrence rewritten as P (16xK f16) x E^T (Kx16 f16)
// WMMA GEMM with E = exp(trans) loop-invariant in VGPRs; running score kept
// in registers in WMMA C/D fragment layout. B=256 -> 16 blocks of 16 batches,
// 8 wave32 per block, one 16-state i-tile per wave, sequential scan over S.

typedef __attribute__((ext_vector_type(16))) _Float16 v16h;
typedef __attribute__((ext_vector_type(8)))  _Float16 v8h;
typedef __attribute__((ext_vector_type(8)))  float    v8f;
typedef __attribute__((ext_vector_type(4)))  float    v4f;

#define SOS_IDX 0
#define EOS_IDX 1
#define NEGV   (-10000.0f)
#define LOG2E  1.4426950408889634f
#define LN2    0.6931471805599453f

constexpr int Bc = 256, Sc = 1024, Tc = 128;
constexpr int SPITCH = Tc + 4;   // float pitch for score rows (bank-conflict padding)
constexpr int PPITCH = Tc + 8;   // f16 pitch for P rows (272B stride, 16B aligned)

__global__ __launch_bounds__(256)
void CRF_scan_wmma_kernel(const float* __restrict__ h,
                          const float* __restrict__ mask,
                          const float* __restrict__ trans,
                          float* __restrict__ out)
{
    __shared__ float                 score[16 * SPITCH]; // row-major copy for P-phase
    __shared__ _Float16              pbuf [16 * PPITCH]; // p = exp(score - rowmax), f16
    __shared__ __align__(16) float   rm   [16];          // rowmax for current step

    const int tid  = threadIdx.x;
    const int lane = tid & 31;
    const int w    = tid >> 5;        // wave id == i-tile id (0..7)
    const int half = lane >> 4;       // ISA K-layout half select
    const int l16  = lane & 15;
    const int b0   = blockIdx.x * 16;
    const long long hrow = (long long)Sc * Tc;

    // ---- Loop-invariant B fragments: B[k=j][n=i] = exp(trans[i][j]) --------
    v16h bfrag[4];
    {
        const int i = w * 16 + l16;
        #pragma unroll
        for (int kc = 0; kc < 4; ++kc) {
            v16h f;
            #pragma unroll
            for (int q = 0; q < 16; ++q) {
                int kl = (q < 8 ? q : q + 8) + 8 * half;
                f[q] = (_Float16)__builtin_amdgcn_exp2f(trans[i * Tc + kc * 32 + kl] * LOG2E);
            }
            bfrag[kc] = f;
        }
    }

    // ---- Init: score0 = NEG except SOS column; register copy in D layout ---
    const int icol = w * 16 + l16;       // this lane's state column
    float sreg[8];                        // score[(r + 8*half)][icol]
    #pragma unroll
    for (int r = 0; r < 8; ++r) {
        sreg[r] = (icol == SOS_IDX) ? 0.0f : NEGV;
        score[(r + 8 * half) * SPITCH + icol] = sreg[r];
    }

    for (int t = 0; t < Sc; ++t) {
        // ---- issue h / mask loads early (no LDS dependency) ----------------
        float hv[8];
        float mv[8];
        #pragma unroll
        for (int r = 0; r < 8; ++r) {
            const int m = r + 8 * half;
            hv[r] = h[(long long)(b0 + m) * hrow + (long long)t * Tc + icol];
            mv[r] = mask[(long long)(b0 + m) * Sc + t];   // 0.0 or 1.0
        }

        __syncthreads();   // prev step's score writes visible

        // ---- P-phase: exact rowmax via half-wave butterfly, then p -> f16 --
        {
            const int m  = tid >> 4;          // row handled by this thread
            const int j0 = (tid & 15) * 8;
            float sv[8];
            #pragma unroll
            for (int q = 0; q < 8; ++q) sv[q] = score[m * SPITCH + j0 + q];
            float rmax = sv[0];
            #pragma unroll
            for (int q = 1; q < 8; ++q) rmax = fmaxf(rmax, sv[q]);
            // 16 threads sharing row m form one half-wave -> xor butterfly
            rmax = fmaxf(rmax, __shfl_xor(rmax, 1, 32));
            rmax = fmaxf(rmax, __shfl_xor(rmax, 2, 32));
            rmax = fmaxf(rmax, __shfl_xor(rmax, 4, 32));
            rmax = fmaxf(rmax, __shfl_xor(rmax, 8, 32));
            if ((tid & 15) == 0) rm[m] = rmax;
            v8h pk;
            #pragma unroll
            for (int q = 0; q < 8; ++q)
                pk[q] = (_Float16)__builtin_amdgcn_exp2f((sv[q] - rmax) * LOG2E);
            *(v8h*)(pbuf + m * PPITCH + j0) = pk;
            // prefetch next timestep's h tile
            if (t + 1 < Sc)
                __builtin_prefetch(
                    &h[(long long)(b0 + m) * hrow + (long long)(t + 1) * Tc + j0], 0, 3);
        }
        __syncthreads();   // pbuf + rm ready

        // ---- A fragments: batch all LDS reads, then two WMMA chains --------
        v16h a[4];
        #pragma unroll
        for (int kc = 0; kc < 4; ++kc) {
            const _Float16* base = pbuf + l16 * PPITCH + kc * 32 + 8 * half;
            v8h lo = *(const v8h*)(base);
            v8h hi = *(const v8h*)(base + 16);
            a[kc] = __builtin_shufflevector(lo, hi,
                      0,1,2,3,4,5,6,7,8,9,10,11,12,13,14,15);
        }
        v8f z = {0.f,0.f,0.f,0.f,0.f,0.f,0.f,0.f};
        v8f acc0 = __builtin_amdgcn_wmma_f32_16x16x32_f16(false, a[0], false, bfrag[0], (short)0, z,    false, false);
        v8f acc1 = __builtin_amdgcn_wmma_f32_16x16x32_f16(false, a[1], false, bfrag[1], (short)0, z,    false, false);
        acc0     = __builtin_amdgcn_wmma_f32_16x16x32_f16(false, a[2], false, bfrag[2], (short)0, acc0, false, false);
        acc1     = __builtin_amdgcn_wmma_f32_16x16x32_f16(false, a[3], false, bfrag[3], (short)0, acc1, false, false);

        // rowmax values for this lane's 8 rows: two aligned vector LDS reads
        v4f rma = *(const v4f*)(rm + 8 * half);
        v4f rmb = *(const v4f*)(rm + 8 * half + 4);
        float rmv[8];
        #pragma unroll
        for (int q = 0; q < 4; ++q) { rmv[q] = rma[q]; rmv[4 + q] = rmb[q]; }

        // ---- Epilogue (branchless): new = h + rowmax + ln(acc);
        //      blended = old + mval*(new - old)  == new*m + old*(1-m) --------
        #pragma unroll
        for (int r = 0; r < 8; ++r) {
            float se  = fmaxf(acc0[r] + acc1[r], 1e-30f);  // single v_max_num
            float neu = hv[r] + rmv[r] + __builtin_amdgcn_logf(se) * LN2;
            float blended = __builtin_fmaf(mv[r], neu - sreg[r], sreg[r]);
            sreg[r] = blended;
            score[(r + 8 * half) * SPITCH + icol] = blended;  // wave-exclusive cols
        }
    }

    __syncthreads();

    // ---- Final: out[b] = logsumexp_i(score[b][i] + trans[EOS][i]) ----------
    if (tid < 16) {
        const int m = tid;
        float mx = -3.4e38f;
        for (int i2 = 0; i2 < Tc; ++i2)
            mx = fmaxf(mx, score[m * SPITCH + i2] + trans[EOS_IDX * Tc + i2]);
        float s = 0.0f;
        for (int i2 = 0; i2 < Tc; ++i2)
            s += __builtin_amdgcn_exp2f(
                     (score[m * SPITCH + i2] + trans[EOS_IDX * Tc + i2] - mx) * LOG2E);
        out[b0 + m] = mx + __builtin_amdgcn_logf(s) * LN2;
    }
}

extern "C" void kernel_launch(void* const* d_in, const int* in_sizes, int n_in,
                              void* d_out, int out_size, void* d_ws, size_t ws_size,
                              hipStream_t stream) {
    (void)in_sizes; (void)n_in; (void)d_ws; (void)ws_size; (void)out_size;
    const float* h     = (const float*)d_in[0];   // (B, S, T) f32
    const float* mask  = (const float*)d_in[1];   // (B, S)    f32
    const float* trans = (const float*)d_in[2];   // (T, T)    f32
    float* out = (float*)d_out;                   // (B,)      f32

    dim3 grid(Bc / 16);   // 16 workgroups, one 16-batch tile each
    dim3 block(256);      // 8 wave32 per workgroup
    CRF_scan_wmma_kernel<<<grid, block, 0, stream>>>(h, mask, trans, out);
}
